// BifurcationAttention_81509889344060
// MI455X (gfx1250) — matrix operand order repaired and checked by reading, not compiled, not run on hardware
//
#include <hip/hip_runtime.h>

typedef __attribute__((ext_vector_type(16))) _Float16 v16h;
typedef __attribute__((ext_vector_type(8)))  _Float16 v8h;
typedef __attribute__((ext_vector_type(8)))  float    v8f;

constexpr int BB = 2, SS = 2048, DD = 512, HH = 8, HDD = 64;

static __device__ inline v16h combine16(v8h lo, v8h hi) {
    v16h r;
#pragma unroll
    for (int i = 0; i < 8; ++i) { r[i] = lo[i]; r[i + 8] = hi[i]; }
    return r;
}

static __device__ inline float hashnorm(unsigned int idx) {
    unsigned a = idx * 747796405u + 2891336453u;
    a = ((a >> ((a >> 28) + 4u)) ^ a) * 277803737u; a ^= a >> 22;
    unsigned b = (idx ^ 0x9E3779B9u) * 747796405u + 2891336453u;
    b = ((b >> ((b >> 28) + 4u)) ^ b) * 277803737u; b ^= b >> 22;
    float u1 = (float)(a >> 8) * (1.0f / 16777216.0f) + 1e-7f;
    float u2 = (float)(b >> 8) * (1.0f / 16777216.0f);
    return __fsqrt_rn(-2.0f * __logf(u1)) * __cosf(6.28318530717958f * u2);
}

// ---------------------------------------------------------------------------
// Kernel 0: bandwidth-bound fp32 -> fp16 down-convert (once per tensor).
// ---------------------------------------------------------------------------
__global__ __launch_bounds__(256) void cvt_f16_kernel(
    const float* __restrict__ src, _Float16* __restrict__ dst, int n)
{
    const int i = (blockIdx.x * 256 + threadIdx.x) * 8;
    if (i >= n) return;
    float4 a = *(const float4*)(src + i);
    float4 b = *(const float4*)(src + i + 4);
    v8h h;
    h[0]=(_Float16)a.x; h[1]=(_Float16)a.y; h[2]=(_Float16)a.z; h[3]=(_Float16)a.w;
    h[4]=(_Float16)b.x; h[5]=(_Float16)b.y; h[6]=(_Float16)b.z; h[7]=(_Float16)b.w;
    *(v8h*)(dst + i) = h;
}

// ---------------------------------------------------------------------------
// Kernel 1: fused QKV projection, pure-f16 NT GEMM (x16 @ W16^T + b).
// Every A/B fragment is exactly two global_load_b128 (no VALU in MAC loop).
// Q,K stored [B,H,S,HD] f16 (Q pre-scaled by 1/sqrt(HD)); V stored
// transposed [B,H,HD,S] f16.  Wave = 16(M) x 64(N) tile.
// ---------------------------------------------------------------------------
__global__ __launch_bounds__(128) void qkv_proj_kernel(
    const _Float16* __restrict__ xh, const _Float16* __restrict__ wh,
    const float* __restrict__ bq, const float* __restrict__ bk,
    const float* __restrict__ bv,
    _Float16* __restrict__ qh, _Float16* __restrict__ kh,
    _Float16* __restrict__ vth)
{
    const int tid  = threadIdx.x;
    const int lane = tid & 31;
    const int wave = tid >> 5;
    const int job  = blockIdx.x * 4 + wave;     // 6144 jobs
    const int mtile = job & 255;                // 256 M tiles over B*S
    const int nj    = job >> 8;                 // 0..23
    const int mat   = nj >> 3;                  // 0:q 1:k 2:v
    const int n0    = (nj & 7) * 64;

    const float* bias = (mat == 0) ? bq : ((mat == 1) ? bk : bv);
    const _Float16* Wm = wh + (size_t)mat * DD * DD;

    const int m0   = mtile * 16;
    const int mrow = lane & 15;
    const int aoff = (lane & 16) ? 16 : 0;      // A-fragment K sub-offset
    const int boff = (lane & 16) ? 16 : 0;      // B-fragment K sub-offset
    const _Float16* xr = xh + (size_t)(m0 + mrow) * DD;

    v8f acc[4] = {};
    for (int kk = 0; kk < DD; kk += 32) {
        v16h A = combine16(*(const v8h*)(xr + kk + aoff),
                           *(const v8h*)(xr + kk + aoff + 8));
#pragma unroll
        for (int nt = 0; nt < 4; ++nt) {
            const _Float16* wr = Wm + (size_t)(n0 + nt * 16 + mrow) * DD + kk + boff;
            v16h Bf = combine16(*(const v8h*)wr, *(const v8h*)(wr + 8));
            acc[nt] = __builtin_amdgcn_wmma_f32_16x16x32_f16(
                false, A, false, Bf, (short)0, acc[nt], false, false);
        }
    }

    const int bidx  = m0 >> 11;       // / S
    const int srow0 = m0 & (SS - 1);
    const int h     = n0 >> 6;
    const int moff  = (lane & 16) ? 8 : 0;
#pragma unroll
    for (int nt = 0; nt < 4; ++nt) {
        const int hd = nt * 16 + mrow;
        const float bval = bias[n0 + hd];
#pragma unroll
        for (int r = 0; r < 8; ++r) {
            const int s = srow0 + r + moff;
            const float val = acc[nt][r] + bval;
            if (mat == 0) {
                qh[((bidx * HH + h) * SS + s) * HDD + hd] = (_Float16)(val * 0.125f);
            } else if (mat == 1) {
                kh[((bidx * HH + h) * SS + s) * HDD + hd] = (_Float16)val;
            } else {
                vth[((bidx * HH + h) * HDD + hd) * SS + s] = (_Float16)val;
            }
        }
    }
}

// ---------------------------------------------------------------------------
// Kernel 2: flash attention.  One wave owns a 16-query tile of one head and
// streams 32 keys/iter: 4 score WMMAs (QK^T) + online softmax + 4 PV WMMAs
// + 1 "P @ ones" WMMA that accumulates the softmax row-sums on the matrix
// unit (deletes the 32-shuffle sum-reduction chain per chunk).
// P tile goes C-layout -> A-layout through a 1KB/wave LDS staging buffer.
// ---------------------------------------------------------------------------
__global__ __launch_bounds__(128) void flash_attn_kernel(
    const _Float16* __restrict__ qh, const _Float16* __restrict__ kh,
    const _Float16* __restrict__ vth, const float* __restrict__ bifp,
    float* __restrict__ out)
{
    __shared__ __align__(16) _Float16 Pld[4][16][32];

    const int tid  = threadIdx.x;
    const int lane = tid & 31;
    const int wave = tid >> 5;
    const int job  = blockIdx.x * 4 + wave;   // 2048 jobs
    const int qtile = job & 127;
    const int bh    = job >> 7;               // b*H + h
    const int q0    = qtile * 16;

    // bifurcation noise coefficient (uniform over grid)
    const float p  = bifp[0];
    const float rr = 1.0f / (1.0f + __expf(-p));
    const float bf = rr * __sinf(3.14159265358979f * rr);
    const float coeff = (fabsf(bf - 0.5f) < 0.1f) ? 0.05f : 0.0f;

    const int mrow = lane & 15;
    const int ao   = (lane & 16) ? 16 : 0;
    const int bo   = (lane & 16) ? 16 : 0;
    const int moff = (lane & 16) ? 8 : 0;

    // constant all-ones B fragment for the row-sum WMMA
    v16h Ones;
#pragma unroll
    for (int i = 0; i < 16; ++i) Ones[i] = (_Float16)1.0f;

    // Q fragments for this query tile (persist in registers), scale pre-folded
    const _Float16* qr = qh + ((size_t)(bh * SS + q0 + mrow)) * HDD;
    v16h Qf[2];
#pragma unroll
    for (int f = 0; f < 2; ++f)
        Qf[f] = combine16(*(const v8h*)(qr + f * 32 + ao),
                          *(const v8h*)(qr + f * 32 + ao + 8));

    const _Float16* kbase = kh + (size_t)bh * SS * HDD;
    const _Float16* vbase = vth + (size_t)bh * HDD * SS;

    float mprev[8];
#pragma unroll
    for (int r = 0; r < 8; ++r) mprev[r] = -__builtin_inff();
    v8f ctx[4] = {};
    v8f ctxsum = {};            // running softmax row-sums (all N columns equal)

    for (int kc = 0; kc < SS; kc += 32) {
        if (kc + 32 < SS)
            __builtin_prefetch(kbase + (size_t)(kc + 32 + mrow) * HDD, 0, 1);

        // ---- scores: two 16x16 tiles over keys [kc,kc+32) -----------------
        v8f sc[2] = {};
#pragma unroll
        for (int st = 0; st < 2; ++st) {
            const _Float16* kr = kbase + (size_t)(kc + st * 16 + mrow) * HDD;
#pragma unroll
            for (int f = 0; f < 2; ++f) {
                v16h Bf = combine16(*(const v8h*)(kr + f * 32 + bo),
                                    *(const v8h*)(kr + f * 32 + bo + 8));
                sc[st] = __builtin_amdgcn_wmma_f32_16x16x32_f16(
                    false, Qf[f], false, Bf, (short)0, sc[st], false, false);
            }
        }

        if (coeff > 0.0f) {
#pragma unroll
            for (int st = 0; st < 2; ++st)
#pragma unroll
                for (int r = 0; r < 8; ++r) {
                    const unsigned key = kc + st * 16 + mrow;
                    const unsigned qq  = q0 + r + moff;
                    const unsigned idx = (((unsigned)bh * SS + qq) << 11) + key;
                    sc[st][r] += coeff * hashnorm(idx);
                }
        }

        // ---- online softmax: only the row-MAX needs a shuffle reduction ---
        float alpha[8];
#pragma unroll
        for (int r = 0; r < 8; ++r) {
            float t = fmaxf(sc[0][r], sc[1][r]);
#pragma unroll
            for (int off = 1; off < 16; off <<= 1)
                t = fmaxf(t, __shfl_xor(t, off, 32));
            const float mn = fmaxf(mprev[r], t);
            alpha[r] = __expf(mprev[r] - mn);
            mprev[r] = mn;
            sc[0][r] = __expf(sc[0][r] - mn);
            sc[1][r] = __expf(sc[1][r] - mn);
        }
#pragma unroll
        for (int nt = 0; nt < 4; ++nt)
#pragma unroll
            for (int r = 0; r < 8; ++r) ctx[nt][r] *= alpha[r];
#pragma unroll
        for (int r = 0; r < 8; ++r) ctxsum[r] *= alpha[r];

        // ---- P tile: C-layout -> LDS row-major -> A-layout fragment -------
#pragma unroll
        for (int st = 0; st < 2; ++st)
#pragma unroll
            for (int r = 0; r < 8; ++r)
                Pld[wave][r + moff][st * 16 + mrow] = (_Float16)sc[st][r];

        v16h Pf = combine16(*(const v8h*)&Pld[wave][mrow][ao],
                            *(const v8h*)&Pld[wave][mrow][ao + 8]);

        // ---- row-sums on the matrix unit: ctxsum += P @ ones --------------
        ctxsum = __builtin_amdgcn_wmma_f32_16x16x32_f16(
            false, Pf, false, Ones, (short)0, ctxsum, false, false);

        // ---- context += P @ V  (V transposed: contiguous key axis) --------
#pragma unroll
        for (int nt = 0; nt < 4; ++nt) {
            const _Float16* vr = vbase + (size_t)(nt * 16 + mrow) * SS + kc + bo;
            v16h Bf = combine16(*(const v8h*)vr, *(const v8h*)(vr + 8));
            ctx[nt] = __builtin_amdgcn_wmma_f32_16x16x32_f16(
                false, Pf, false, Bf, (short)0, ctx[nt], false, false);
        }
    }

    // ---- epilogue: divide by row sums, write [B,S,D] f32 -------------------
    const int b = bh >> 3, h = bh & 7;
    float inv[8];
#pragma unroll
    for (int r = 0; r < 8; ++r) inv[r] = 1.0f / ctxsum[r];
#pragma unroll
    for (int nt = 0; nt < 4; ++nt)
#pragma unroll
        for (int r = 0; r < 8; ++r) {
            const int s = q0 + r + moff;
            out[((size_t)(b * SS + s)) * DD + h * HDD + nt * 16 + mrow] =
                ctx[nt][r] * inv[r];
        }
}

// ---------------------------------------------------------------------------
extern "C" void kernel_launch(void* const* d_in, const int* in_sizes, int n_in,
                              void* d_out, int out_size, void* d_ws, size_t ws_size,
                              hipStream_t stream) {
    const float* x   = (const float*)d_in[0];
    const float* Wq  = (const float*)d_in[1];
    const float* bq  = (const float*)d_in[2];
    const float* Wk  = (const float*)d_in[3];
    const float* bk  = (const float*)d_in[4];
    const float* Wv  = (const float*)d_in[5];
    const float* bv  = (const float*)d_in[6];
    const float* bif = (const float*)d_in[7];

    const size_t N  = (size_t)BB * HH * SS * HDD;   // 2,097,152 elems
    const size_t NW = (size_t)DD * DD;              //   262,144 elems
    _Float16* qh  = (_Float16*)d_ws;
    _Float16* kh  = qh + N;
    _Float16* vth = kh + N;
    _Float16* xh  = vth + N;
    _Float16* wh  = xh + N;                         // Wq | Wk | Wv (f16)

    // one-time fp32 -> fp16 down-converts (L2-resident afterwards)
    cvt_f16_kernel<<<(int)(N / (256 * 8)), 256, 0, stream>>>(x, xh, (int)N);
    cvt_f16_kernel<<<(int)(NW / (256 * 8)), 256, 0, stream>>>(Wq, wh, (int)NW);
    cvt_f16_kernel<<<(int)(NW / (256 * 8)), 256, 0, stream>>>(Wk, wh + NW, (int)NW);
    cvt_f16_kernel<<<(int)(NW / (256 * 8)), 256, 0, stream>>>(Wv, wh + 2 * NW, (int)NW);

    // 6144 wave-jobs, 4 waves (128 threads) per block
    qkv_proj_kernel<<<1536, 128, 0, stream>>>(xh, wh, bq, bk, bv, qh, kh, vth);
    // 2048 wave-jobs
    flash_attn_kernel<<<512, 128, 0, stream>>>(qh, kh, vth, bif,
                                               (float*)d_out);
}
